// set_upconv_module_62062277427559
// MI455X (gfx1250) — matrix operand order repaired and checked by reading, not compiled, not run on hardware
//
#include <hip/hip_runtime.h>

typedef _Float16 f16;
typedef __attribute__((ext_vector_type(16))) _Float16 v16h;
typedef __attribute__((ext_vector_type(8)))  _Float16 v8h;
typedef __attribute__((ext_vector_type(4)))  _Float16 v4h;
typedef __attribute__((ext_vector_type(8)))  float    v8f;

#define BB   2
#define NN   16384
#define SS   4096
#define KNB  8
#define D1c  64
#define D2c  128
#define EPSc 1e-5f

#define ROWS1 (BB * NN * KNB)   /* 262144 rows for layer-1 GEMMs */
#define ROWS2 (BB * NN)         /* 32768  rows for layer-2 GEMM  */

extern __shared__ __align__(32) char smem[];

// ---------------------------------------------------------------------------
// Zero the stats accumulators (harness poisons ws; must re-zero every call).
// ---------------------------------------------------------------------------
__global__ __launch_bounds__(256) void zero_kernel(float* p, int n) {
  int i = blockIdx.x * 256 + threadIdx.x;
  if (i < n) p[i] = 0.f;
}

// ---------------------------------------------------------------------------
// Brute-force 8-NN. One block = 256 queries of one batch; xyz2 staged in LDS
// padded to float4 so each candidate is a single ds_load_b128 broadcast.
// ---------------------------------------------------------------------------
__global__ __launch_bounds__(256) void knn_kernel(const float* __restrict__ xyz1,
                                                  const float* __restrict__ xyz2,
                                                  int* __restrict__ idxbuf) {
  float* sh = (float*)smem;                     // SS*4 floats = 64 KB
  const int tid  = threadIdx.x;
  const int bpb  = NN / 256;
  const int b    = blockIdx.x / bpb;
  const int qloc = (blockIdx.x % bpb) * 256 + tid;

  const size_t sbase = (size_t)b * SS * 3;
  for (int s = tid; s < SS; s += 256) {
    sh[4 * s + 0] = xyz2[sbase + 3 * s + 0];
    sh[4 * s + 1] = xyz2[sbase + 3 * s + 1];
    sh[4 * s + 2] = xyz2[sbase + 3 * s + 2];
    sh[4 * s + 3] = 0.f;
  }
  __syncthreads();

  const size_t q = (size_t)b * NN + qloc;
  const float qx = xyz1[q * 3 + 0];
  const float qy = xyz1[q * 3 + 1];
  const float qz = xyz1[q * 3 + 2];

  float d0 = 3.4e38f, d1 = 3.4e38f, d2 = 3.4e38f, d3 = 3.4e38f,
        d4 = 3.4e38f, d5 = 3.4e38f, d6 = 3.4e38f, d7 = 3.4e38f;
  int   i0 = 0, i1 = 0, i2 = 0, i3 = 0, i4 = 0, i5 = 0, i6 = 0, i7 = 0;

#define CSWAP(A, B)                                                     \
  if (d##A < d##B) { float td = d##A; d##A = d##B; d##B = td;           \
                     int ti = i##A; i##A = i##B; i##B = ti; }

  for (int s = 0; s < SS; ++s) {
    float4 bp = *(const float4*)(sh + 4 * s);
    float dx = qx - bp.x, dy = qy - bp.y, dz = qz - bp.z;
    float d  = fmaf(dx, dx, fmaf(dy, dy, dz * dz));
    if (d < d7) {
      d7 = d; i7 = s;
      CSWAP(7, 6) CSWAP(6, 5) CSWAP(5, 4) CSWAP(4, 3)
      CSWAP(3, 2) CSWAP(2, 1) CSWAP(1, 0)
    }
  }
#undef CSWAP
  int* o = idxbuf + q * KNB;
  o[0] = i0; o[1] = i1; o[2] = i2; o[3] = i3;
  o[4] = i4; o[5] = i5; o[6] = i6; o[7] = i7;
}

// ---------------------------------------------------------------------------
// GEMM1a: rows = gathered [feat2 | xyz_diff] (131, padded to 160) x w1a(->128)
// Block: 128 rows, 8 waves x 16 rows. Stores f16 pre-act + channel stats.
// ---------------------------------------------------------------------------
__global__ __launch_bounds__(256) void gemm1a_kernel(
    const float* __restrict__ xyz1, const float* __restrict__ xyz2,
    const float* __restrict__ feat2, const float* __restrict__ w1a,
    const float* __restrict__ b1a, const int* __restrict__ idxbuf,
    f16* __restrict__ h1, float* __restrict__ gsum, float* __restrict__ gsq) {
  f16*   ldsA  = (f16*)smem;                  // 128 x 160 f16 = 40960 B
  f16*   ldsWt = (f16*)(smem + 40960);        // 128(n) x 160(k) = 40960 B
  float* sSum  = (float*)(smem + 81920);      // 128
  float* sSq   = sSum + 128;                  // 128
  const int tid     = threadIdx.x;
  const int rowBase = blockIdx.x * 128;

  if (tid < 128) { sSum[tid] = 0.f; sSq[tid] = 0.f; }

  // Stage w1a transposed (n-major) as f16, zero-padded K:131->160 (v4h stores)
  for (int i = tid; i < 128 * 40; i += 256) {
    int n = i / 40, k4 = (i - n * 40) * 4;
    v4h h;
#pragma unroll
    for (int j = 0; j < 4; ++j) {
      int k = k4 + j;
      h[j] = (f16)((k < 131) ? w1a[k * 128 + n] : 0.f);
    }
    *(v4h*)(ldsWt + n * 160 + k4) = h;
  }
  // Stage gathered A rows (2 threads per row), packed v4h LDS stores
  {
    int r    = tid >> 1, part = tid & 1;
    int grow = rowBase + r;
    int p    = grow >> 3;                 // point = b*N + n
    int b    = p / NN;
    int n    = p - b * NN;
    int s    = idxbuf[grow];
    const float* f2 = feat2 + ((size_t)(b * SS + s)) * D2c;
    f16* dst = ldsA + r * 160;
    if (part == 0) {
      for (int c = 0; c < 80; c += 4) {
        float4 v = *(const float4*)(f2 + c);
        v4h h = { (f16)v.x, (f16)v.y, (f16)v.z, (f16)v.w };
        *(v4h*)(dst + c) = h;
      }
    } else {
      for (int c = 80; c < 128; c += 4) {
        float4 v = *(const float4*)(f2 + c);
        v4h h = { (f16)v.x, (f16)v.y, (f16)v.z, (f16)v.w };
        *(v4h*)(dst + c) = h;
      }
      const float* p2 = xyz2 + ((size_t)(b * SS + s)) * 3;
      const float* p1 = xyz1 + ((size_t)(b * NN + n)) * 3;
      v4h t = { (f16)(p2[0] - p1[0]), (f16)(p2[1] - p1[1]),
                (f16)(p2[2] - p1[2]), (f16)0.f };
      *(v4h*)(dst + 128) = t;
      v4h z = {};
      for (int c = 132; c < 160; c += 4) *(v4h*)(dst + c) = z;
    }
  }
  __syncthreads();

  const int lane = tid & 31, wid = tid >> 5;
  const int m = lane & 15, half = lane >> 4;
  const int rB = wid * 16;

  v8f acc[8] = {};
  const f16* aRow = ldsA + (rB + m) * 160;
#pragma unroll
  for (int ks = 0; ks < 5; ++ks) {
    v8h alo = *(const v8h*)(aRow + ks * 32 + half * 8);
    v8h ahi = *(const v8h*)(aRow + ks * 32 + 16 + half * 8);
    v16h a  = __builtin_shufflevector(alo, ahi, 0, 1, 2, 3, 4, 5, 6, 7,
                                      8, 9, 10, 11, 12, 13, 14, 15);
#pragma unroll
    for (int nt = 0; nt < 8; ++nt) {
      const f16* bRow = ldsWt + (nt * 16 + m) * 160 + ks * 32;
      v8h blo = *(const v8h*)(bRow + half * 8);
      v8h bhi = *(const v8h*)(bRow + 16 + half * 8);
      v16h bf = __builtin_shufflevector(blo, bhi, 0, 1, 2, 3, 4, 5, 6, 7,
                                        8, 9, 10, 11, 12, 13, 14, 15);
      acc[nt] = __builtin_amdgcn_wmma_f32_16x16x32_f16(
          false, a, false, bf, (short)0, acc[nt], false, false);
    }
  }

#pragma unroll
  for (int nt = 0; nt < 8; ++nt) {
    int   col  = nt * 16 + m;
    float bias = b1a[col];
    float ls = 0.f, lq = 0.f;
#pragma unroll
    for (int r = 0; r < 8; ++r) {
      int   grow = rowBase + rB + half * 8 + r;
      float v    = acc[nt][r] + bias;
      ls += v; lq += v * v;
      h1[(size_t)grow * 128 + col] = (f16)v;
    }
    atomicAdd(&sSum[col], ls);
    atomicAdd(&sSq[col], lq);
  }
  __syncthreads();
  if (tid < 128) { atomicAdd(&gsum[tid], sSum[tid]); atomicAdd(&gsq[tid], sSq[tid]); }
}

// ---------------------------------------------------------------------------
// Per-channel BN finalize: scale = gamma*rsqrt(var+eps), shift = beta - mean*scale
// ---------------------------------------------------------------------------
__global__ void bn_finalize_kernel(const float* sum, const float* sq,
                                   const float* gamma, const float* beta,
                                   float* scale, float* shift, int C, float ic) {
  int c = threadIdx.x;
  if (c < C) {
    float m  = sum[c] * ic;
    float v  = sq[c] * ic - m * m;
    float rs = rsqrtf(v + EPSc) * gamma[c];
    scale[c] = rs;
    shift[c] = beta[c] - m * rs;
  }
}

// ---------------------------------------------------------------------------
// GEMM1b fused with K-maxpool: BN1+ReLU applied while staging h1 (vectorized
// v8h loads/stores), 128->64 GEMM, channel stats from accumulators, per-point
// pooled max & min stored. Each lane's 8 accumulator rows == 8 neighbors of
// exactly one point, so pooling is register-local.
// ---------------------------------------------------------------------------
__global__ __launch_bounds__(256) void gemm1b_kernel(
    const f16* __restrict__ h1, const float* __restrict__ w1b,
    const float* __restrict__ b1b, const float* __restrict__ scale1,
    const float* __restrict__ shift1, float* __restrict__ pmax,
    float* __restrict__ pmin, float* __restrict__ gsum, float* __restrict__ gsq) {
  f16*   ldsA  = (f16*)smem;                 // 128 x 128 f16 = 32768 B
  f16*   ldsWt = (f16*)(smem + 32768);       // 64 x 128 f16 = 16384 B
  float* sSum  = (float*)(smem + 49152);     // 64
  float* sSq   = sSum + 64;
  const int tid = threadIdx.x;
  const int rowBase = blockIdx.x * 128;

  if (tid < 64) { sSum[tid] = 0.f; sSq[tid] = 0.f; }
  for (int i = tid; i < 64 * 32; i += 256) {
    int n = i >> 5, k4 = (i & 31) * 4;
    v4h h;
#pragma unroll
    for (int j = 0; j < 4; ++j) h[j] = (f16)w1b[(k4 + j) * 64 + n];
    *(v4h*)(ldsWt + n * 128 + k4) = h;
  }
  for (int i = tid; i < 128 * 16; i += 256) {
    int r = i >> 4, c8 = (i & 15) * 8;
    v8h hv = *(const v8h*)(h1 + (size_t)(rowBase + r) * 128 + c8);
    v8h o;
#pragma unroll
    for (int j = 0; j < 8; ++j) {
      int c = c8 + j;
      float v = (float)hv[j];
      o[j] = (f16)fmaxf(v * scale1[c] + shift1[c], 0.f);
    }
    *(v8h*)(ldsA + r * 128 + c8) = o;
  }
  __syncthreads();

  const int lane = tid & 31, wid = tid >> 5;
  const int m = lane & 15, half = lane >> 4;
  const int rB = wid * 16;

  v8f acc[4] = {};
  const f16* aRow = ldsA + (rB + m) * 128;
#pragma unroll
  for (int ks = 0; ks < 4; ++ks) {
    v8h alo = *(const v8h*)(aRow + ks * 32 + half * 8);
    v8h ahi = *(const v8h*)(aRow + ks * 32 + 16 + half * 8);
    v16h a  = __builtin_shufflevector(alo, ahi, 0, 1, 2, 3, 4, 5, 6, 7,
                                      8, 9, 10, 11, 12, 13, 14, 15);
#pragma unroll
    for (int nt = 0; nt < 4; ++nt) {
      const f16* bRow = ldsWt + (nt * 16 + m) * 128 + ks * 32;
      v8h blo = *(const v8h*)(bRow + half * 8);
      v8h bhi = *(const v8h*)(bRow + 16 + half * 8);
      v16h bf = __builtin_shufflevector(blo, bhi, 0, 1, 2, 3, 4, 5, 6, 7,
                                        8, 9, 10, 11, 12, 13, 14, 15);
      acc[nt] = __builtin_amdgcn_wmma_f32_16x16x32_f16(
          false, a, false, bf, (short)0, acc[nt], false, false);
    }
  }

  const int p = (rowBase >> 3) + wid * 2 + half;   // point index for this lane
#pragma unroll
  for (int nt = 0; nt < 4; ++nt) {
    int   c    = nt * 16 + m;
    float bias = b1b[c];
    float ls = 0.f, lq = 0.f, mx = -3.4e38f, mn = 3.4e38f;
#pragma unroll
    for (int r = 0; r < 8; ++r) {
      float v = acc[nt][r] + bias;
      ls += v; lq += v * v;
      mx = fmaxf(mx, v); mn = fminf(mn, v);
    }
    pmax[(size_t)p * 64 + c] = mx;
    pmin[(size_t)p * 64 + c] = mn;
    atomicAdd(&sSum[c], ls);
    atomicAdd(&sSq[c], lq);
  }
  __syncthreads();
  if (tid < 64) { atomicAdd(&gsum[tid], sSum[tid]); atomicAdd(&gsq[tid], sSq[tid]); }
}

// ---------------------------------------------------------------------------
// GEMM2a: rows = [BN2+ReLU(pooled) | feat1] (128) x w2a(->64). Sign-aware
// pooled pick (max if BN scale>=0 else min) makes pool/BN commute exactly.
// ---------------------------------------------------------------------------
__global__ __launch_bounds__(256) void gemm2a_kernel(
    const float* __restrict__ feat1, const float* __restrict__ w2a,
    const float* __restrict__ b2a, const float* __restrict__ pmax,
    const float* __restrict__ pmin, const float* __restrict__ scale2,
    const float* __restrict__ shift2, float* __restrict__ pre3,
    float* __restrict__ gsum, float* __restrict__ gsq) {
  f16*   ldsA  = (f16*)smem;                 // 128 x 128
  f16*   ldsWt = (f16*)(smem + 32768);       // 64 x 128
  float* sSum  = (float*)(smem + 49152);
  float* sSq   = sSum + 64;
  const int tid = threadIdx.x;
  const int rowBase = blockIdx.x * 128;      // rows are points here

  if (tid < 64) { sSum[tid] = 0.f; sSq[tid] = 0.f; }
  for (int i = tid; i < 64 * 32; i += 256) {
    int n = i >> 5, k4 = (i & 31) * 4;
    v4h h;
#pragma unroll
    for (int j = 0; j < 4; ++j) h[j] = (f16)w2a[(k4 + j) * 64 + n];
    *(v4h*)(ldsWt + n * 128 + k4) = h;
  }
  for (int i = tid; i < 128 * 32; i += 256) {
    int r = i >> 5, c4 = (i & 31) * 4;
    size_t p = (size_t)(rowBase + r);
    v4h o;
    if (c4 < 64) {
      float4 mx = *(const float4*)(pmax + p * 64 + c4);
      float4 mn = *(const float4*)(pmin + p * 64 + c4);
      float mxa[4] = { mx.x, mx.y, mx.z, mx.w };
      float mna[4] = { mn.x, mn.y, mn.z, mn.w };
#pragma unroll
      for (int j = 0; j < 4; ++j) {
        float s  = scale2[c4 + j];
        float bs = (s >= 0.f) ? mxa[j] : mna[j];
        o[j] = (f16)fmaxf(s * bs + shift2[c4 + j], 0.f);
      }
    } else {
      float4 v = *(const float4*)(feat1 + p * 64 + (c4 - 64));
      o = (v4h){ (f16)v.x, (f16)v.y, (f16)v.z, (f16)v.w };
    }
    *(v4h*)(ldsA + r * 128 + c4) = o;
  }
  __syncthreads();

  const int lane = tid & 31, wid = tid >> 5;
  const int m = lane & 15, half = lane >> 4;
  const int rB = wid * 16;

  v8f acc[4] = {};
  const f16* aRow = ldsA + (rB + m) * 128;
#pragma unroll
  for (int ks = 0; ks < 4; ++ks) {
    v8h alo = *(const v8h*)(aRow + ks * 32 + half * 8);
    v8h ahi = *(const v8h*)(aRow + ks * 32 + 16 + half * 8);
    v16h a  = __builtin_shufflevector(alo, ahi, 0, 1, 2, 3, 4, 5, 6, 7,
                                      8, 9, 10, 11, 12, 13, 14, 15);
#pragma unroll
    for (int nt = 0; nt < 4; ++nt) {
      const f16* bRow = ldsWt + (nt * 16 + m) * 128 + ks * 32;
      v8h blo = *(const v8h*)(bRow + half * 8);
      v8h bhi = *(const v8h*)(bRow + 16 + half * 8);
      v16h bf = __builtin_shufflevector(blo, bhi, 0, 1, 2, 3, 4, 5, 6, 7,
                                        8, 9, 10, 11, 12, 13, 14, 15);
      acc[nt] = __builtin_amdgcn_wmma_f32_16x16x32_f16(
          false, a, false, bf, (short)0, acc[nt], false, false);
    }
  }

#pragma unroll
  for (int nt = 0; nt < 4; ++nt) {
    int   c    = nt * 16 + m;
    float bias = b2a[c];
    float ls = 0.f, lq = 0.f;
#pragma unroll
    for (int r = 0; r < 8; ++r) {
      int   grow = rowBase + rB + half * 8 + r;
      float v    = acc[nt][r] + bias;
      ls += v; lq += v * v;
      pre3[(size_t)grow * 64 + c] = v;
    }
    atomicAdd(&sSum[c], ls);
    atomicAdd(&sSq[c], lq);
  }
  __syncthreads();
  if (tid < 64) { atomicAdd(&gsum[tid], sSum[tid]); atomicAdd(&gsq[tid], sSq[tid]); }
}

// ---------------------------------------------------------------------------
// Final BN + ReLU into d_out.
// ---------------------------------------------------------------------------
__global__ __launch_bounds__(256) void bnrelu_out_kernel(
    const float* __restrict__ pre3, const float* __restrict__ scale3,
    const float* __restrict__ shift3, float* __restrict__ out) {
  int i = blockIdx.x * 256 + threadIdx.x;
  int c = i & 63;
  out[i] = fmaxf(pre3[i] * scale3[c] + shift3[c], 0.f);
}

// ---------------------------------------------------------------------------
extern "C" void kernel_launch(void* const* d_in, const int* in_sizes, int n_in,
                              void* d_out, int out_size, void* d_ws, size_t ws_size,
                              hipStream_t stream) {
  const float* xyz1  = (const float*)d_in[0];
  const float* xyz2  = (const float*)d_in[1];
  const float* feat1 = (const float*)d_in[2];
  const float* feat2 = (const float*)d_in[3];
  const float* w1a   = (const float*)d_in[4];
  const float* b1a   = (const float*)d_in[5];
  const float* g1a   = (const float*)d_in[6];
  const float* bb1a  = (const float*)d_in[7];
  const float* w1b   = (const float*)d_in[8];
  const float* b1b   = (const float*)d_in[9];
  const float* g1b   = (const float*)d_in[10];
  const float* bb1b  = (const float*)d_in[11];
  const float* w2a   = (const float*)d_in[12];
  const float* b2a   = (const float*)d_in[13];
  const float* g2a   = (const float*)d_in[14];
  const float* bb2a  = (const float*)d_in[15];

  char* ws = (char*)d_ws;
  int*   idxbuf = (int*)ws;                                //  1 MB
  f16*   h1     = (f16*)(ws + (1ull << 20));               // 64 MB
  float* pmax   = (float*)(ws + 68157440ull);              //  8 MB
  float* pmin   = (float*)(ws + 76546048ull);              //  8 MB
  float* pre3   = (float*)(ws + 84934656ull);              //  8 MB
  float* stats  = (float*)(ws + 93323264ull);              //  4 KB
  float* gsum1 = stats;        float* gsq1 = stats + 128;
  float* gsum2 = stats + 256;  float* gsq2 = stats + 320;
  float* gsum3 = stats + 384;  float* gsq3 = stats + 448;
  float* scale1 = stats + 512; float* shift1 = stats + 640;
  float* scale2 = stats + 768; float* shift2 = stats + 832;
  float* scale3 = stats + 896; float* shift3 = stats + 960;

  zero_kernel<<<2, 256, 0, stream>>>(stats, 512);
  knn_kernel<<<BB * (NN / 256), 256, SS * 4 * 4, stream>>>(xyz1, xyz2, idxbuf);
  gemm1a_kernel<<<ROWS1 / 128, 256, 82944, stream>>>(xyz1, xyz2, feat2, w1a, b1a,
                                                     idxbuf, h1, gsum1, gsq1);
  bn_finalize_kernel<<<1, 128, 0, stream>>>(gsum1, gsq1, g1a, bb1a, scale1, shift1,
                                            128, 1.0f / (float)ROWS1);
  gemm1b_kernel<<<ROWS1 / 128, 256, 49664, stream>>>(h1, w1b, b1b, scale1, shift1,
                                                     pmax, pmin, gsum2, gsq2);
  bn_finalize_kernel<<<1, 64, 0, stream>>>(gsum2, gsq2, g1b, bb1b, scale2, shift2,
                                           64, 1.0f / (float)ROWS1);
  gemm2a_kernel<<<ROWS2 / 128, 256, 49664, stream>>>(feat1, w2a, b2a, pmax, pmin,
                                                     scale2, shift2, pre3, gsum3, gsq3);
  bn_finalize_kernel<<<1, 64, 0, stream>>>(gsum3, gsq3, g2a, bb2a, scale3, shift3,
                                           64, 1.0f / (float)ROWS2);
  bnrelu_out_kernel<<<(ROWS2 * 64) / 256, 256, 0, stream>>>(pre3, scale3, shift3,
                                                            (float*)d_out);
}
